// Net_49323404427684
// MI455X (gfx1250) — compile-verified, matrix operand order checked by reading, not run on hardware
//
#include <hip/hip_runtime.h>
#include <hip/hip_bf16.h>
#include <math.h>

typedef __attribute__((ext_vector_type(16))) _Float16 v16h;
typedef __attribute__((ext_vector_type(8)))  float    v8f;

#define N_NODES   100000
#define N_EDGES   1600000
#define N_GRAPHS  256
#define DIM       32
#define NC        2
#define TILES     (N_NODES / 16)   /* 6250, even -> 2 waves/block, no tail */
#define BN_EPS    1e-5f

// ---------------------------------------------------------------------------
// A/B fragment K-index helper for v_wmma_f32_16x16x32_f16 (16-bit, 16x32 A):
// lanes 0-15: K in {0..7} (VGPR0-3) and {16..23} (VGPR4-7)
// lanes 16-31: K in {8..15} and {24..31}; each VGPR packs (k, k+1).
// ---------------------------------------------------------------------------
__device__ __forceinline__ int a_k_of(int lane, int v) {
  int kb = (lane >> 4) * 8;
  return (v < 4) ? (kb + 2 * v) : (16 + kb + 2 * (v - 4));
}

// ---------------------------------------------------------------------------
// Edge scatter-add: agg[dst[e]*D + c] += h[src[e]*D + c].
// Consecutive lanes share an edge -> coalesced gather + coalesced atomics.
// ---------------------------------------------------------------------------
template <int D>
__global__ __launch_bounds__(256)
void scatter_add_kernel(const float* __restrict__ h,
                        const int* __restrict__ src,
                        const int* __restrict__ dst,
                        float* __restrict__ agg, int total) {
  int tid = blockIdx.x * blockDim.x + threadIdx.x;
  if (tid >= total) return;
  int e = tid / D;
  int c = tid % D;
  atomicAdd(&agg[dst[e] * D + c], h[src[e] * D + c]);
}

// ---------------------------------------------------------------------------
// Layer-1 fused MLP (NF=2 input): hidden = relu(z @ w_in + b_in) computed
// directly in A-fragment layout (2 FMAs per element), then WMMA for the
// 32x32 second GEMM. Emits y (pre-BN) and BN column sum / sumsq atomically.
// One wave == one 16-node tile. 2 waves per block.
// ---------------------------------------------------------------------------
__global__ __launch_bounds__(64)
void gin_mlp1_kernel(const float* __restrict__ x, const float* __restrict__ agg,
                     const float* __restrict__ w_in, const float* __restrict__ b_in,
                     const float* __restrict__ w_out, const float* __restrict__ b_out,
                     float* __restrict__ y, float* __restrict__ stats) {
  int lane = threadIdx.x & 31;
  int wv   = threadIdx.x >> 5;
  int m0   = (blockIdx.x * 2 + wv) * 16;
  int row  = m0 + (lane & 15);

  float z0 = x[row * 2 + 0] + agg[row * 2 + 0];
  float z1 = x[row * 2 + 1] + agg[row * 2 + 1];

  v16h a;
#pragma unroll
  for (int v = 0; v < 8; ++v) {
    int k = a_k_of(lane, v);
    float h0 = fmaf(z0, w_in[k],      fmaf(z1, w_in[DIM + k],      b_in[k]));
    float h1 = fmaf(z0, w_in[k + 1],  fmaf(z1, w_in[DIM + k + 1],  b_in[k + 1]));
    a[2 * v]     = (_Float16)(h0 > 0.f ? h0 : 0.f);
    a[2 * v + 1] = (_Float16)(h1 > 0.f ? h1 : 0.f);
  }

  int colb  = lane & 15;
  int rbase = (lane >> 4) * 8;
#pragma unroll
  for (int t = 0; t < 2; ++t) {
    int col = colb + 16 * t;
    v16h b;
#pragma unroll
    for (int v = 0; v < 8; ++v) {
      int k = a_k_of(lane, v);
      b[2 * v]     = (_Float16)w_out[k * DIM + col];
      b[2 * v + 1] = (_Float16)w_out[(k + 1) * DIM + col];
    }
    v8f c = {};
    c = __builtin_amdgcn_wmma_f32_16x16x32_f16(false, a, false, b, (short)0, c,
                                               false, false);
    float bias = b_out[col];
    float s = 0.f, s2 = 0.f;
#pragma unroll
    for (int r = 0; r < 8; ++r) {
      float val = c[r] + bias;
      val = val > 0.f ? val : 0.f;
      y[(m0 + rbase + r) * DIM + col] = val;
      s += val; s2 += val * val;
    }
    atomicAdd(&stats[col], s);
    atomicAdd(&stats[DIM + col], s2);
  }
}

// ---------------------------------------------------------------------------
// Layers 2..5 fused MLP: z = h + agg (f32 -> f16 A-frag), two chained 32x32
// GEMMs via v_wmma_f32_16x16x32_f16, D->A relayout through per-wave LDS.
// Emits y (pre-BN) + BN column stats.
// ---------------------------------------------------------------------------
__global__ __launch_bounds__(64)
void gin_mlp_kernel(const float* __restrict__ h, const float* __restrict__ agg,
                    const float* __restrict__ w_in, const float* __restrict__ b_in,
                    const float* __restrict__ w_out, const float* __restrict__ b_out,
                    float* __restrict__ y, float* __restrict__ stats) {
  __shared__ float lds[2][16][DIM + 1];
  int lane = threadIdx.x & 31;
  int wv   = threadIdx.x >> 5;
  int m0   = (blockIdx.x * 2 + wv) * 16;
  int row  = m0 + (lane & 15);

  // --- A fragment of z = h + agg ---
  v16h a;
#pragma unroll
  for (int v = 0; v < 8; ++v) {
    int k = a_k_of(lane, v);
    a[2 * v]     = (_Float16)(h[row * DIM + k]     + agg[row * DIM + k]);
    a[2 * v + 1] = (_Float16)(h[row * DIM + k + 1] + agg[row * DIM + k + 1]);
  }

  int colb  = lane & 15;
  int rbase = (lane >> 4) * 8;

  // --- stage 1: hidden = relu(z @ w_in + b_in), park in LDS for relayout ---
#pragma unroll
  for (int t = 0; t < 2; ++t) {
    int col = colb + 16 * t;
    v16h b;
#pragma unroll
    for (int v = 0; v < 8; ++v) {
      int k = a_k_of(lane, v);
      b[2 * v]     = (_Float16)w_in[k * DIM + col];
      b[2 * v + 1] = (_Float16)w_in[(k + 1) * DIM + col];
    }
    v8f c = {};
    c = __builtin_amdgcn_wmma_f32_16x16x32_f16(false, a, false, b, (short)0, c,
                                               false, false);
    float bias = b_in[col];
#pragma unroll
    for (int r = 0; r < 8; ++r) {
      float val = c[r] + bias;
      lds[wv][rbase + r][col] = val > 0.f ? val : 0.f;
    }
  }
  __syncthreads();

  // --- relayout hidden D-layout -> A fragment ---
  v16h a2;
#pragma unroll
  for (int v = 0; v < 8; ++v) {
    int k = a_k_of(lane, v);
    a2[2 * v]     = (_Float16)lds[wv][lane & 15][k];
    a2[2 * v + 1] = (_Float16)lds[wv][lane & 15][k + 1];
  }

  // --- stage 2: y = relu(hidden @ w_out + b_out) + BN stats ---
#pragma unroll
  for (int t = 0; t < 2; ++t) {
    int col = colb + 16 * t;
    v16h b;
#pragma unroll
    for (int v = 0; v < 8; ++v) {
      int k = a_k_of(lane, v);
      b[2 * v]     = (_Float16)w_out[k * DIM + col];
      b[2 * v + 1] = (_Float16)w_out[(k + 1) * DIM + col];
    }
    v8f c = {};
    c = __builtin_amdgcn_wmma_f32_16x16x32_f16(false, a2, false, b, (short)0, c,
                                               false, false);
    float bias = b_out[col];
    float s = 0.f, s2 = 0.f;
#pragma unroll
    for (int r = 0; r < 8; ++r) {
      float val = c[r] + bias;
      val = val > 0.f ? val : 0.f;
      y[(m0 + rbase + r) * DIM + col] = val;
      s += val; s2 += val * val;
    }
    atomicAdd(&stats[col], s);
    atomicAdd(&stats[DIM + col], s2);
  }
}

// ---------------------------------------------------------------------------
// BatchNorm apply: h = gamma*(y-mean)*rsqrt(var+eps)+beta (biased var).
// ---------------------------------------------------------------------------
__global__ __launch_bounds__(256)
void bn_kernel(const float* __restrict__ y, const float* __restrict__ stats,
               const float* __restrict__ gamma, const float* __restrict__ beta,
               float* __restrict__ h) {
  int tid = blockIdx.x * blockDim.x + threadIdx.x;
  if (tid >= N_NODES * DIM) return;
  int c = tid & (DIM - 1);
  float invn = 1.0f / (float)N_NODES;
  float mean = stats[c] * invn;
  float var  = stats[DIM + c] * invn - mean * mean;
  float sc   = gamma[c] * rsqrtf(var + BN_EPS);
  h[tid] = sc * (y[tid] - mean) + beta[c];
}

// ---------------------------------------------------------------------------
// global_add_pool: g[batch[i]][c] += h[i][c]
// ---------------------------------------------------------------------------
__global__ __launch_bounds__(256)
void pool_kernel(const float* __restrict__ h, const int* __restrict__ batch,
                 float* __restrict__ g) {
  int tid = blockIdx.x * blockDim.x + threadIdx.x;
  if (tid >= N_NODES * DIM) return;
  int i = tid >> 5;
  int c = tid & (DIM - 1);
  atomicAdd(&g[batch[i] * DIM + c], h[tid]);
}

// ---------------------------------------------------------------------------
// Head: relu(g @ fc1 + b) @ fc2 + b -> log_softmax. One thread per graph;
// trivial FLOPs (256 x 32 x 32), not worth a GEMM path.
// ---------------------------------------------------------------------------
__global__ __launch_bounds__(N_GRAPHS)
void head_kernel(const float* __restrict__ g,
                 const float* __restrict__ fc1_w, const float* __restrict__ fc1_b,
                 const float* __restrict__ fc2_w, const float* __restrict__ fc2_b,
                 float* __restrict__ out) {
  int r = threadIdx.x;
  float gi[DIM];
#pragma unroll
  for (int c = 0; c < DIM; ++c) gi[c] = g[r * DIM + c];

  float logit[NC];
#pragma unroll
  for (int k = 0; k < NC; ++k) logit[k] = fc2_b[k];

  for (int j = 0; j < DIM; ++j) {
    float hj = fc1_b[j];
#pragma unroll
    for (int c = 0; c < DIM; ++c) hj = fmaf(gi[c], fc1_w[c * DIM + j], hj);
    hj = hj > 0.f ? hj : 0.f;
#pragma unroll
    for (int k = 0; k < NC; ++k) logit[k] = fmaf(hj, fc2_w[j * NC + k], logit[k]);
  }
  float m = logit[0] > logit[1] ? logit[0] : logit[1];
  float lse = m + logf(expf(logit[0] - m) + expf(logit[1] - m));
#pragma unroll
  for (int k = 0; k < NC; ++k) out[r * NC + k] = logit[k] - lse;
}

// ---------------------------------------------------------------------------
extern "C" void kernel_launch(void* const* d_in, const int* in_sizes, int n_in,
                              void* d_out, int out_size, void* d_ws, size_t ws_size,
                              hipStream_t stream) {
  (void)in_sizes; (void)n_in; (void)out_size; (void)ws_size;
  const float* x      = (const float*)d_in[0];
  const int*   ei     = (const int*)d_in[1];
  const int*   batch  = (const int*)d_in[2];
  const float* w1_in  = (const float*)d_in[3];
  const float* b1_in  = (const float*)d_in[4];
  const float* w1_out = (const float*)d_in[5];
  const float* b1_out = (const float*)d_in[6];
  const float* ws_in  = (const float*)d_in[7];
  const float* bs_in  = (const float*)d_in[8];
  const float* ws_out = (const float*)d_in[9];
  const float* bs_out = (const float*)d_in[10];
  const float* gamma  = (const float*)d_in[11];
  const float* beta   = (const float*)d_in[12];
  const float* fc1_w  = (const float*)d_in[13];
  const float* fc1_b  = (const float*)d_in[14];
  const float* fc2_w  = (const float*)d_in[15];
  const float* fc2_b  = (const float*)d_in[16];
  float* out = (float*)d_out;

  const int* srcv = ei;
  const int* dstv = ei + N_EDGES;

  float* h   = (float*)d_ws;                       // N x 32
  float* yb  = h   + (size_t)N_NODES * DIM;        // N x 32 (pre-BN)
  float* agg = yb  + (size_t)N_NODES * DIM;        // N x 32 (scatter target)
  float* st  = agg + (size_t)N_NODES * DIM;        // 64 (sum | sumsq)
  float* gp  = st  + 64;                           // G x 32

  // ---- layer 1 (NF = 2) ----
  hipMemsetAsync(agg, 0, (size_t)N_NODES * 2 * sizeof(float), stream);
  hipMemsetAsync(st,  0, 2 * DIM * sizeof(float), stream);
  {
    int total = N_EDGES * 2;
    scatter_add_kernel<2><<<(total + 255) / 256, 256, 0, stream>>>(
        x, srcv, dstv, agg, total);
  }
  gin_mlp1_kernel<<<TILES / 2, 64, 0, stream>>>(x, agg, w1_in, b1_in,
                                                w1_out, b1_out, yb, st);
  bn_kernel<<<(N_NODES * DIM + 255) / 256, 256, 0, stream>>>(
      yb, st, gamma, beta, h);

  // ---- layers 2..5 (DIM = 32) ----
  for (int l = 0; l < 4; ++l) {
    hipMemsetAsync(agg, 0, (size_t)N_NODES * DIM * sizeof(float), stream);
    hipMemsetAsync(st,  0, 2 * DIM * sizeof(float), stream);
    int total = N_EDGES * DIM;
    scatter_add_kernel<DIM><<<(total + 255) / 256, 256, 0, stream>>>(
        h, srcv, dstv, agg, total);
    gin_mlp_kernel<<<TILES / 2, 64, 0, stream>>>(
        h, agg, ws_in + l * DIM * DIM, bs_in + l * DIM,
        ws_out + l * DIM * DIM, bs_out + l * DIM, yb, st);
    bn_kernel<<<(N_NODES * DIM + 255) / 256, 256, 0, stream>>>(
        yb, st, gamma + (l + 1) * DIM, beta + (l + 1) * DIM, h);
  }

  // ---- pooling + head ----
  hipMemsetAsync(gp, 0, (size_t)N_GRAPHS * DIM * sizeof(float), stream);
  pool_kernel<<<(N_NODES * DIM + 255) / 256, 256, 0, stream>>>(h, batch, gp);
  head_kernel<<<1, N_GRAPHS, 0, stream>>>(gp, fc1_w, fc1_b, fc2_w, fc2_b, out);
}
